// StackedRNNModel_80229989089928
// MI455X (gfx1250) — matrix operand order repaired and checked by reading, not compiled, not run on hardware
//
#include <hip/hip_runtime.h>
#include <hip/hip_bf16.h>
#include <stdint.h>

// ---------------------------------------------------------------------------
// Problem constants (reference: B=64, S=2048, D=128, H=256, O=1, 3 layers)
// ---------------------------------------------------------------------------
#define BB   64
#define SS   2048
#define DD   128
#define HH   256
#define NWG  16      // workgroups in the persistent LSTM kernel
#define LSTM_THREADS 128  // 4 waves of 32

typedef __attribute__((ext_vector_type(16))) __bf16        v16bf;
typedef __attribute__((ext_vector_type(8)))  float         v8f;
typedef __attribute__((ext_vector_type(4)))  unsigned int  u32x4;
typedef __attribute__((ext_vector_type(8)))  unsigned int  u32x8;
typedef __attribute__((ext_vector_type(4)))  int           i32x4;
typedef __attribute__((ext_vector_type(8)))  int           i32x8;

union Frag {
    v16bf bf;
    struct Pair { u32x4 lo; u32x4 hi; } q;
};

__device__ __forceinline__ unsigned short f2bf(float f) {
    unsigned u = __builtin_bit_cast(unsigned, f);
    unsigned r = u + 0x7FFFu + ((u >> 16) & 1u);   // round-to-nearest-even
    return (unsigned short)(r >> 16);
}
__device__ __forceinline__ float bf2f(unsigned short h) {
    return __builtin_bit_cast(float, (unsigned)h << 16);
}
__device__ __forceinline__ float sigmoidf_(float x) {
    return 1.0f / (1.0f + __expf(-x));
}

// ---------------------------------------------------------------------------
// Prep kernels
// ---------------------------------------------------------------------------
__global__ void init_ctrl_kernel(unsigned int* ctrl) {
    if (threadIdx.x < 16) ctrl[threadIdx.x] = 0u;
}

__global__ void convert_x_kernel(const float* __restrict__ X,
                                 unsigned short* __restrict__ Xbf, int n) {
    int i = blockIdx.x * blockDim.x + threadIdx.x;
    if (i < n) Xbf[i] = f2bf(X[i]);
}

// Build per-WG permuted weight block:  wuout[wg][lcol][k]  (bf16, column-major
// over K so B-fragments are contiguous), lcol = gate*16 + unit_in_wg.
__global__ void prep_w_kernel(const float* __restrict__ W,   // [Din][4H]
                              const float* __restrict__ U,   // [H][4H]
                              const float* __restrict__ b,   // [4H]
                              int Din,
                              unsigned short* __restrict__ wuout,
                              float* __restrict__ bout) {
    const int K = Din + HH;
    const int total = NWG * 64 * K;
    for (int e = blockIdx.x * blockDim.x + threadIdx.x; e < total;
         e += gridDim.x * blockDim.x) {
        int k  = e % K;
        int lc = (e / K) % 64;
        int wg = e / (K * 64);
        int g = lc >> 4, j = lc & 15;
        int c = g * HH + wg * 16 + j;              // global column in 4H
        float v = (k < Din) ? W[(size_t)k * (4 * HH) + c]
                            : U[(size_t)(k - Din) * (4 * HH) + c];
        wuout[e] = f2bf(v);
        if (k == 0) bout[wg * 64 + lc] = b[c];
    }
}

// ---------------------------------------------------------------------------
// Persistent LSTM layer kernel (grid = NWG blocks x 128 threads).
// Each WG owns 16 hidden units (64 gate columns). Weights live in LDS,
// loaded once via the Tensor Data Mover. Per step: one 64x64x(KX+256) bf16
// WMMA GEMM + gate math; cell state stays in VGPRs; grid barrier per step.
// ---------------------------------------------------------------------------
template <int KX>
__global__ __launch_bounds__(LSTM_THREADS) void lstm_layer_kernel(
    const unsigned short* __restrict__ xin,   // bf16 [B][S][KX]
    const unsigned short* __restrict__ wug,   // bf16 [NWG][64][KTOT]
    const float* __restrict__ biasg,          // f32  [NWG][64]
    const long long* __restrict__ Tlen,       // i64  [B]
    unsigned short* __restrict__ hout,        // bf16 [B][S][H]
    unsigned int* bar)                        // grid-barrier counter
{
    constexpr int KTOT = KX + HH;             // 384 or 512
    __shared__ unsigned short wu[64 * KTOT];  // 48KB / 64KB

    const int wg   = blockIdx.x;
    const int tid  = threadIdx.x;
    const int lane = tid & 31;
    const int wave = tid >> 5;
    const int nlo  = lane & 15;               // N index inside a 16x16 tile
    const int khi  = lane >> 4;               // lane-half select

    // ---- Load this WG's contiguous 64*KTOT bf16 weight chunk into LDS ----
#if __has_builtin(__builtin_amdgcn_tensor_load_to_lds)
    {
        const unsigned NE = 64u * (unsigned)KTOT;          // elements (2B each)
        unsigned lds_off  = (unsigned)(uintptr_t)(&wu[0]);
        uint64_t ga = (uint64_t)(uintptr_t)(wug + (size_t)wg * 64 * KTOT);
        // D# group 0: count=1 | lds_addr | global_addr | type=2
        u32x4 g0 = { 1u,
                     lds_off,
                     (unsigned)(ga & 0xFFFFFFFFu),
                     (unsigned)((ga >> 32) & 0x01FFFFFFu) | 0x80000000u };
        // D# group 1: data_size=2B, 1-D tensor: dim0=NE, dim1=1, tile0=NE
        i32x8 g1 = { (int)0x00010000u,
                     (int)((NE & 0xFFFFu) << 16),          // tensor_dim0[15:0]
                     (int)(1u << 16),                      // tensor_dim1 = 1
                     (int)((NE & 0xFFFFu) << 16),          // tile_dim0
                     0,
                     (int)NE,                              // dim0_stride low
                     0, 0 };
        i32x4 gz4 = { 0, 0, 0, 0 };
        if (tid < 32) {   // one wave issues the TDM op (EXEC-independent DMA)
#if defined(__clang_major__) && (__clang_major__ >= 23)
            i32x8 gz8 = { 0, 0, 0, 0, 0, 0, 0, 0 };
            __builtin_amdgcn_tensor_load_to_lds(g0, g1, gz4, gz4, gz8, 0);
#else
            __builtin_amdgcn_tensor_load_to_lds(g0, g1, gz4, gz4, 0);
#endif
            __builtin_amdgcn_s_wait_tensorcnt(0);
        }
    }
#else
    for (int i = tid * 8; i < 64 * KTOT; i += LSTM_THREADS * 8) {
        *(u32x4*)&wu[i] = *(const u32x4*)&wug[(size_t)wg * 64 * KTOT + i];
    }
#endif
    __syncthreads();

    // ---- per-lane loop invariants ----
    const int mrow = wave * 16 + nlo;         // batch row this lane loads (A)
    const int ucol = wg * 16 + nlo;           // hidden unit this lane owns (D)
    const float bi = biasg[wg * 64 + 0 * 16 + nlo];
    const float bf = biasg[wg * 64 + 1 * 16 + nlo];
    const float bc = biasg[wg * 64 + 2 * 16 + nlo];
    const float bo = biasg[wg * 64 + 3 * 16 + nlo];
    int tcap[8];
#pragma unroll
    for (int v = 0; v < 8; ++v)
        tcap[v] = (int)Tlen[wave * 16 + v + 8 * khi];   // batch of D row v

    v8f cst = {0.f,0.f,0.f,0.f,0.f,0.f,0.f,0.f};        // carried cell state
    v8f hst = {0.f,0.f,0.f,0.f,0.f,0.f,0.f,0.f};        // carried hidden state

    for (int t = 0; t < SS; ++t) {
        v8f acc[4];
#pragma unroll
        for (int g = 0; g < 4; ++g)
            acc[g] = (v8f){0.f,0.f,0.f,0.f,0.f,0.f,0.f,0.f};

        const unsigned short* xr = xin  + ((size_t)mrow * SS + t) * KX;
        const unsigned short* hr = hout + ((size_t)mrow * SS + (t - 1)) * HH;

#pragma unroll
        for (int kk = 0; kk < KTOT; kk += 32) {
            // A fragment (16x32 bf16): per-lane two 16-B chunks, ISA layout
            Frag a;
            if (kk < KX) {
                a.q.lo = *(const u32x4*)(xr + kk + khi * 8);
                a.q.hi = *(const u32x4*)(xr + kk + 16 + khi * 8);
            } else if (t > 0) {
                const int kh = kk - KX;
                a.q.lo = *(const u32x4*)(hr + kh + khi * 8);
                a.q.hi = *(const u32x4*)(hr + kh + 16 + khi * 8);
            } else {
                a.q.lo = (u32x4){0u,0u,0u,0u};
                a.q.hi = (u32x4){0u,0u,0u,0u};
            }
            // 4 gate B fragments from LDS (32x16 bf16, K-contiguous per lane)
#pragma unroll
            for (int g = 0; g < 4; ++g) {
                Frag b;
                const unsigned short* bp =
                    &wu[(size_t)(g * 16 + nlo) * KTOT + kk + khi * 8];
                b.q.lo = *(const u32x4*)bp;
                b.q.hi = *(const u32x4*)(bp + 16);
                acc[g] = __builtin_amdgcn_wmma_f32_16x16x32_bf16(
                    false, a.bf, false, b.bf, (short)0, acc[g], false, false);
            }
        }

        // ---- element-wise LSTM cell (Keras gate order i,f,g,o) ----
#pragma unroll
        for (int v = 0; v < 8; ++v) {
            const int batch = wave * 16 + v + 8 * khi;
            const float gi = sigmoidf_(acc[0][v] + bi);
            const float gf = sigmoidf_(acc[1][v] + bf);
            const float gc = tanhf   (acc[2][v] + bc);
            const float go = sigmoidf_(acc[3][v] + bo);
            const float cn = gf * cst[v] + gi * gc;
            const float hn = go * tanhf(cn);
            const bool  mt = t < tcap[v];
            cst[v] = mt ? cn : cst[v];
            hst[v] = mt ? hn : hst[v];
            hout[((size_t)batch * SS + t) * HH + ucol] = f2bf(hst[v]);
        }

        if (t + 1 < SS) __builtin_prefetch(xr + KX, 0, 1);  // next x_t row

        // ---- grid barrier: all WGs publish h_t before anyone reads it ----
        __syncthreads();
        if (tid == 0) {
            __threadfence();
            __hip_atomic_fetch_add(bar, 1u, __ATOMIC_RELEASE,
                                   __HIP_MEMORY_SCOPE_AGENT);
            const unsigned target = (unsigned)(t + 1) * NWG;
            while (__hip_atomic_load(bar, __ATOMIC_ACQUIRE,
                                     __HIP_MEMORY_SCOPE_AGENT) < target) {
                __builtin_amdgcn_s_sleep(1);
            }
        }
        __syncthreads();
        __threadfence();
    }
}

// ---------------------------------------------------------------------------
// Output head: out[b,s] = sigmoid(h2[b,s,:] . Wo + bo)   (one wave per row)
// ---------------------------------------------------------------------------
__global__ void out_head_kernel(const unsigned short* __restrict__ h2, // bf16
                                const float* __restrict__ Wo,          // [H]
                                const float* __restrict__ bo,          // [1]
                                float* __restrict__ out) {
    const int w    = blockIdx.x * (blockDim.x >> 5) + (threadIdx.x >> 5);
    const int lane = threadIdx.x & 31;
    const unsigned short* row = h2 + (size_t)w * HH;
    float s = 0.f;
    for (int k = lane; k < HH; k += 32) s += bf2f(row[k]) * Wo[k];
#pragma unroll
    for (int off = 16; off > 0; off >>= 1) s += __shfl_down(s, off, 32);
    if (lane == 0) out[w] = sigmoidf_(s + bo[0]);
}

// ---------------------------------------------------------------------------
// Workspace layout (bytes)
// ---------------------------------------------------------------------------
static constexpr size_t CTRL_OFF = 0;                       // 4 KB ctrl
static constexpr size_t WU_OFF   = 4096;                    // 3 x 1MB slots
static constexpr size_t WU_SLOT  = 1048576;
static constexpr size_t BIAS_OFF = WU_OFF + 3 * WU_SLOT;    // 3 x 4KB
static constexpr size_t XBF_OFF  = BIAS_OFF + 16384;
static constexpr size_t XBF_SZ   = (size_t)BB * SS * DD * 2;    // 33.5 MB
static constexpr size_t HA_OFF   = XBF_OFF + XBF_SZ;
static constexpr size_t HSEQ_SZ  = (size_t)BB * SS * HH * 2;    // 67.1 MB
static constexpr size_t HB_OFF   = HA_OFF + HSEQ_SZ;

extern "C" void kernel_launch(void* const* d_in, const int* in_sizes, int n_in,
                              void* d_out, int out_size, void* d_ws,
                              size_t ws_size, hipStream_t stream) {
    (void)in_sizes; (void)n_in; (void)out_size; (void)ws_size;

    const float*     X  = (const float*)d_in[0];
    const long long* T  = (const long long*)d_in[1];
    const float*     W0 = (const float*)d_in[2];
    const float*     U0 = (const float*)d_in[3];
    const float*     b0 = (const float*)d_in[4];
    const float*     W1 = (const float*)d_in[5];
    const float*     U1 = (const float*)d_in[6];
    const float*     b1 = (const float*)d_in[7];
    const float*     W2 = (const float*)d_in[8];
    const float*     U2 = (const float*)d_in[9];
    const float*     b2 = (const float*)d_in[10];
    const float*     Wo = (const float*)d_in[11];
    const float*     bo = (const float*)d_in[12];
    float*           out = (float*)d_out;

    char* ws = (char*)d_ws;
    unsigned int*   ctrl  = (unsigned int*)(ws + CTRL_OFF);
    unsigned short* wu0   = (unsigned short*)(ws + WU_OFF + 0 * WU_SLOT);
    unsigned short* wu1   = (unsigned short*)(ws + WU_OFF + 1 * WU_SLOT);
    unsigned short* wu2   = (unsigned short*)(ws + WU_OFF + 2 * WU_SLOT);
    float*          bias0 = (float*)(ws + BIAS_OFF + 0 * 4096);
    float*          bias1 = (float*)(ws + BIAS_OFF + 1 * 4096);
    float*          bias2 = (float*)(ws + BIAS_OFF + 2 * 4096);
    unsigned short* xbf   = (unsigned short*)(ws + XBF_OFF);
    unsigned short* ha    = (unsigned short*)(ws + HA_OFF);
    unsigned short* hb    = (unsigned short*)(ws + HB_OFF);

    // prep
    init_ctrl_kernel<<<1, 32, 0, stream>>>(ctrl);
    const int nx = BB * SS * DD;
    convert_x_kernel<<<(nx + 255) / 256, 256, 0, stream>>>(X, xbf, nx);
    prep_w_kernel<<<1024, 256, 0, stream>>>(W0, U0, b0, DD, wu0, bias0);
    prep_w_kernel<<<1024, 256, 0, stream>>>(W1, U1, b1, HH, wu1, bias1);
    prep_w_kernel<<<1024, 256, 0, stream>>>(W2, U2, b2, HH, wu2, bias2);

    // 3 persistent LSTM layers (ping-pong activation buffers)
    lstm_layer_kernel<DD><<<NWG, LSTM_THREADS, 0, stream>>>(
        xbf, wu0, bias0, T, ha, ctrl + 0);
    lstm_layer_kernel<HH><<<NWG, LSTM_THREADS, 0, stream>>>(
        ha, wu1, bias1, T, hb, ctrl + 1);
    lstm_layer_kernel<HH><<<NWG, LSTM_THREADS, 0, stream>>>(
        hb, wu2, bias2, T, ha, ctrl + 2);

    // sigmoid head: B*S rows, 8 waves per 256-thread block
    out_head_kernel<<<(BB * SS) / 8, 256, 0, stream>>>(ha, Wo, bo, out);
}
